// OnlyAttention_2808908611939
// MI455X (gfx1250) — matrix operand order repaired and checked by reading, not compiled, not run on hardware
//
#include <hip/hip_runtime.h>

typedef _Float16 f16;
typedef _Float16 v16h __attribute__((ext_vector_type(16)));
typedef _Float16 v8h  __attribute__((ext_vector_type(8)));
typedef float    v8f  __attribute__((ext_vector_type(8)));
typedef float    v4f  __attribute__((ext_vector_type(4)));
typedef int      v4i  __attribute__((vector_size(16)));   // matches builtin param type

#define N_SHARDS 4
#define N_SEQS   32
#define N_HEADS  32
#define N_KVH    8
#define GQA      4
#define HD       128
#define KVLEN    1024
#define TILE_K   32
#define WAVES_PER_BLOCK 4
#define TILES_PER_WAVE  (KVLEN / TILE_K / WAVES_PER_BLOCK)   // 8
#define KV_ROW   (N_KVH * HD)                                 // floats per (seq,key) row group

// ---- gfx1250 async global->LDS path (guarded; falls back to VGPR staging) ----
#if defined(__HIP_DEVICE_COMPILE__) && defined(__has_builtin)
#if __has_builtin(__builtin_amdgcn_global_load_async_to_lds_b128)
#define ASYNC_V 1
#endif
#endif
#ifndef ASYNC_V
#define ASYNC_V 0
#endif

#define AS1 __attribute__((address_space(1)))
#define AS3 __attribute__((address_space(3)))

#if ASYNC_V
// LDS byte offset == low 32 bits of the generic address (aperture rules).
__device__ __forceinline__ AS3 v4i* to_lds(void* p) {
  return (AS3 v4i*)(unsigned)(unsigned long long)p;
}
__device__ __forceinline__ AS1 v4i* to_glb(const void* p) {
  return (AS1 v4i*)(unsigned long long)p;
}
__device__ __forceinline__ void wait_async0() {
#if __has_builtin(__builtin_amdgcn_s_wait_asynccnt)
  __builtin_amdgcn_s_wait_asynccnt(0);
#else
  asm volatile("s_wait_asynccnt 0x0" ::: "memory");
#endif
  asm volatile("" ::: "memory");   // keep LDS reads below the wait
}
#endif

__device__ __forceinline__ v8f wmma16x16x32(v16h a, v16h b, v8f c) {
  // D = A(16x32 f16) * B(32x16 f16) + C(16x16 f32)
  return __builtin_amdgcn_wmma_f32_16x16x32_f16(false, a, false, b, (short)0, c,
                                                false, false);
}

// A-operand fragment (16x32 f16): lane holds row (lane&15); element e maps to
// K = (e<8 ? e : e+8); +8*hi and +32*c are folded into `p` by the caller.
__device__ __forceinline__ v16h load_a_k(const float* __restrict__ p) {
  const v4f* qp = (const v4f*)p;
  v4f x0 = qp[0], x1 = qp[1], x2 = qp[4], x3 = qp[5];
  v16h a;
#pragma unroll
  for (int i = 0; i < 4; ++i) {
    a[i]      = (f16)x0[i];
    a[i + 4]  = (f16)x1[i];
    a[i + 8]  = (f16)x2[i];
    a[i + 12] = (f16)x3[i];
  }
  return a;
}

// ---------------------------------------------------------------------------
// Kernel 1: per-(shard, seq, kv-head) flash-decode partial attention.
// ---------------------------------------------------------------------------
__global__ __launch_bounds__(128, 1)
void flash_decode_partial(const float* __restrict__ q,
                          const float* __restrict__ k,
                          const float* __restrict__ v,
                          float* __restrict__ part,
                          float* __restrict__ mlog,
                          float* __restrict__ esum) {
#if ASYNC_V
  // 64 KB block: V tiles [w][key][d] f32 during the loop; merge scratch after.
  __shared__ __align__(16) unsigned char smem[WAVES_PER_BLOCK * TILE_K * HD * 4];
  float* const vstG = (float*)smem;
  float* const MACC = (float*)smem;                               // [w][GQA][HD]
  float* const MML  = (float*)smem + WAVES_PER_BLOCK * GQA * HD;  // [w][GQA]
  float* const MLL  = MML + WAVES_PER_BLOCK * GQA;                // [w][GQA]
#else
  __shared__ __align__(16) f16 vstage[WAVES_PER_BLOCK][TILE_K * HD]; // f16 [d][key]
  __shared__ float maccA[WAVES_PER_BLOCK][GQA][HD];
  __shared__ float mmlA[WAVES_PER_BLOCK][GQA];
  __shared__ float mllA[WAVES_PER_BLOCK][GQA];
  float* const MACC = &maccA[0][0][0];
  float* const MML  = &mmlA[0][0];
  float* const MLL  = &mllA[0][0];
#endif

  const int b    = blockIdx.x;         // (s, n, kvh)
  const int s    = b >> 8;
  const int n    = (b >> 3) & 31;
  const int kvh  = b & 7;

  const int tid  = threadIdx.x;
  const int lane = tid & 31;
  const int w    = tid >> 5;
  const int hi   = lane >> 4;          // lane half (wave32)
  const int lo16 = lane & 15;

  const float scale = 0.08838834764831845f;   // 128^-0.5

  // ---- q group as B-operand fragments (cols = heads, K = d-chunk of 32)
  v16h bq[4];
  {
    const float* qb = q + ((size_t)n * N_HEADS + (size_t)kvh * GQA + lo16) * HD;
#pragma unroll
    for (int c = 0; c < 4; ++c) {
      v16h t;
#pragma unroll
      for (int e = 0; e < 16; ++e) t[e] = (f16)0.0f;
      if (lo16 < GQA) {
        const v4f* p = (const v4f*)(qb + c * 32 + hi * 8);
        v4f x0 = p[0], x1 = p[1], x2 = p[4], x3 = p[5];
#pragma unroll
        for (int i = 0; i < 4; ++i) {
          t[i]      = (f16)x0[i];
          t[i + 4]  = (f16)x1[i];
          t[i + 8]  = (f16)x2[i];
          t[i + 12] = (f16)x3[i];
        }
      }
      bq[c] = t;
    }
  }

  const size_t seq_key_base = ((size_t)s * N_SEQS + n) * KVLEN;  // key-row index
  const size_t kvh_off      = (size_t)kvh * HD;

  v8f zf = {};
  v8f acc[8];
#pragma unroll
  for (int j = 0; j < 8; ++j) acc[j] = zf;

  float m_run = -__builtin_inff();
  float l_run = 0.0f;

  for (int it = 0; it < TILES_PER_WAVE; ++it) {
    const int kbase = (w * TILES_PER_WAVE + it) * TILE_K;

    if (it + 1 < TILES_PER_WAVE)
      __builtin_prefetch(k + (seq_key_base + kbase + TILE_K + lane) * (size_t)KV_ROW + kvh_off, 0, 1);

#if ASYNC_V
    // ---- DMA V tile (32 keys x 128 f32, row-major) straight into LDS.
    // One async b128 per lane per key row: 512 B fully coalesced, zero VALU.
    // Overlaps with the QK loads/WMMAs below; waited before the PV reads.
    {
      const float* gbase = v + (seq_key_base + kbase) * (size_t)KV_ROW + kvh_off + 4 * lane;
      float* lbase = vstG + (size_t)w * TILE_K * HD + 4 * lane;
#pragma unroll 8
      for (int i = 0; i < TILE_K; ++i) {
        __builtin_amdgcn_global_load_async_to_lds_b128(
            to_glb(gbase + (size_t)i * KV_ROW), to_lds(lbase + i * HD), 0, 0);
      }
    }
#else
    // ---- stage V tile into LDS transposed f16 [d][key] via VGPRs
#pragma unroll 4
    for (int i = 0; i < 16; ++i) {
      const float* v0 = v + (seq_key_base + kbase + 2 * i) * (size_t)KV_ROW + kvh_off + 4 * lane;
      const float* v1 = v0 + KV_ROW;
      v4f a  = *(const v4f*)v0;
      v4f bb = *(const v4f*)v1;
#pragma unroll
      for (int qd = 0; qd < 4; ++qd) {
        unsigned short u0 = __builtin_bit_cast(unsigned short, (f16)a[qd]);
        unsigned short u1 = __builtin_bit_cast(unsigned short, (f16)bb[qd]);
        unsigned pk = (unsigned)u0 | ((unsigned)u1 << 16);
        *(unsigned*)&vstage[w][(4 * lane + qd) * TILE_K + 2 * i] = pk;
      }
    }
#endif

    // ---- S^T = K_tile * q^T (two 16-key subtiles, accumulate over d chunks)
    v8f st0 = {}, st1 = {};
    const float* kr0 = k + (seq_key_base + kbase + lo16) * (size_t)KV_ROW + kvh_off + hi * 8;
    const float* kr1 = kr0 + 16 * (size_t)KV_ROW;
#pragma unroll
    for (int c = 0; c < 4; ++c) {
      v16h a0 = load_a_k(kr0 + c * 32);
      v16h a1 = load_a_k(kr1 + c * 32);
      st0 = wmma16x16x32(a0, bq[c], st0);
      st1 = wmma16x16x32(a1, bq[c], st1);
    }

    // ---- online softmax: head = lane&15; pair lanes {lane, lane^16}
    float s0[8], s1[8];
    float tmax = -__builtin_inff();
#pragma unroll
    for (int vv = 0; vv < 8; ++vv) {
      s0[vv] = st0[vv] * scale;
      s1[vv] = st1[vv] * scale;
      tmax = fmaxf(tmax, fmaxf(s0[vv], s1[vv]));
    }
    tmax = fmaxf(tmax, __shfl_xor(tmax, 16, 32));
    const float nm   = fmaxf(m_run, tmax);
    const float corr = __expf(m_run - nm);
    float p0[8], p1[8], tsum = 0.0f;
#pragma unroll
    for (int vv = 0; vv < 8; ++vv) {
      p0[vv] = __expf(s0[vv] - nm);
      p1[vv] = __expf(s1[vv] - nm);
      tsum += p0[vv] + p1[vv];
    }
    l_run = l_run * corr + tsum;
    m_run = nm;

    // S^T D-layout == P A-layout: feeds the PV matmul with no shuffles.
    v16h pA;
#pragma unroll
    for (int e = 0; e < 8; ++e) {
      pA[e]     = (f16)p0[e];
      pA[e + 8] = (f16)p1[e];
    }

    // rescale accumulators (rows 0..3 = live heads)
    const float cb0 = __shfl(corr, 0, 32);
    const float cb1 = __shfl(corr, 1, 32);
    const float cb2 = __shfl(corr, 2, 32);
    const float cb3 = __shfl(corr, 3, 32);
#pragma unroll
    for (int j = 0; j < 8; ++j) {
      acc[j][0] *= cb0; acc[j][1] *= cb1; acc[j][2] *= cb2; acc[j][3] *= cb3;
    }

    // ---- O += P * V_tile
#if ASYNC_V
    wait_async0();
    {
      const float* vb = vstG + (size_t)w * TILE_K * HD;   // [key][d] f32 in LDS
#pragma unroll
      for (int j = 0; j < 8; ++j) {
        const int d = 16 * j + lo16;
        v16h bV;
#pragma unroll
        for (int e = 0; e < 16; ++e)
          bV[e] = (f16)vb[(e + 8 * hi + (e >= 8 ? 8 : 0)) * HD + d];        // B: keys 8*hi+e / 16+8*hi+(e-8)
        acc[j] = wmma16x16x32(pA, bV, acc[j]);
      }
    }
#else
#pragma unroll
    for (int j = 0; j < 8; ++j) {
      const int d = 16 * j + lo16;
      const v8h* pv = (const v8h*)&vstage[w][d * TILE_K + 8 * hi];
      v8h blo = pv[0], bhi = pv[2];
      v16h bV;
#pragma unroll
      for (int e = 0; e < 8; ++e) { bV[e] = blo[e]; bV[e + 8] = bhi[e]; }
      acc[j] = wmma16x16x32(pA, bV, acc[j]);
    }
#endif
  }

  // ---- cross-wave online-softmax merge through LDS
  float l_tot = l_run + __shfl_xor(l_run, 16, 32);
  __syncthreads();                       // everyone done with V staging buffers
  if (hi == 0) {
#pragma unroll
    for (int j = 0; j < 8; ++j)
#pragma unroll
      for (int h = 0; h < GQA; ++h)
        MACC[((size_t)w * GQA + h) * HD + 16 * j + lo16] = acc[j][h];
    if (lo16 < GQA) { MML[w * GQA + lo16] = m_run; MLL[w * GQA + lo16] = l_tot; }
  }
  __syncthreads();

  const int d = tid;                     // 0..127
  const size_t gidx = (((size_t)s * N_SEQS + n) * N_KVH + kvh) * GQA;
#pragma unroll
  for (int h = 0; h < GQA; ++h) {
    float M = -__builtin_inff();
#pragma unroll
    for (int ww = 0; ww < WAVES_PER_BLOCK; ++ww) M = fmaxf(M, MML[ww * GQA + h]);
    float Ls = 0.0f, num = 0.0f;
#pragma unroll
    for (int ww = 0; ww < WAVES_PER_BLOCK; ++ww) {
      const float wt = __expf(MML[ww * GQA + h] - M);
      Ls  += wt * MLL[ww * GQA + h];
      num += wt * MACC[((size_t)ww * GQA + h) * HD + d];
    }
    part[(gidx + h) * HD + d] = num / Ls;
    if (tid == 0) { mlog[gidx + h] = M; esum[gidx + h] = Ls; }
  }
}

// ---------------------------------------------------------------------------
// Kernel 2: gatherlayer — online-softmax combine across shards.
// ---------------------------------------------------------------------------
__global__ __launch_bounds__(128, 1)
void flash_decode_combine(const float* __restrict__ part,
                          const float* __restrict__ mlog,
                          const float* __restrict__ esum,
                          float* __restrict__ out) {
  const int b   = blockIdx.x;            // n*8 + kvh
  const int n   = b >> 3;
  const int kvh = b & 7;
  const int d   = threadIdx.x;
#pragma unroll
  for (int h = 0; h < GQA; ++h) {
    size_t idx[N_SHARDS];
    float  m[N_SHARDS], l[N_SHARDS];
    float gm = -__builtin_inff();
#pragma unroll
    for (int s = 0; s < N_SHARDS; ++s) {
      idx[s] = (((size_t)s * N_SEQS + n) * N_KVH + kvh) * GQA + h;
      m[s] = mlog[idx[s]];
      l[s] = esum[idx[s]];
      gm = fmaxf(gm, m[s]);
    }
    float wsum = 0.0f, num = 0.0f;
#pragma unroll
    for (int s = 0; s < N_SHARDS; ++s) {
      const float wt = __expf(m[s] - gm) * l[s];
      wsum += wt;
      num  += wt * part[idx[s] * HD + d];
    }
    out[((size_t)n * N_HEADS + (size_t)kvh * GQA + h) * HD + d] = num / wsum;
  }
}

extern "C" void kernel_launch(void* const* d_in, const int* in_sizes, int n_in,
                              void* d_out, int out_size, void* d_ws, size_t ws_size,
                              hipStream_t stream) {
  (void)in_sizes; (void)n_in; (void)out_size; (void)ws_size;
  const float* q = (const float*)d_in[0];
  const float* k = (const float*)d_in[1];
  const float* v = (const float*)d_in[2];
  float* out = (float*)d_out;

  float* wsf  = (float*)d_ws;
  float* part = wsf;                                                  // 4*32*8*4*128 f32
  float* mlog = wsf + (size_t)N_SHARDS * N_SEQS * N_KVH * GQA * HD;   // 4096 f32
  float* esum = mlog + (size_t)N_SHARDS * N_SEQS * N_KVH * GQA;       // 4096 f32

  flash_decode_partial<<<N_SHARDS * N_SEQS * N_KVH, 128, 0, stream>>>(q, k, v,
                                                                      part, mlog, esum);
  flash_decode_combine<<<N_SEQS * N_KVH, 128, 0, stream>>>(part, mlog, esum, out);
}
